// PedestrianPredictionFormer_47845935677990
// MI455X (gfx1250) — matrix operand order repaired and checked by reading, not compiled, not run on hardware
//
#include <hip/hip_runtime.h>
#include <hip/hip_bf16.h>
#include <stdint.h>

// ---------------- model dimensions ----------------
#define N_PED 4096
#define OBS   8
#define PRED  12
#define DIM   128
#define PPATCH 4
#define PSTRIDE 2
#define LPATCH 3
#define FEATW 32
#define NHEAD 4
#define DFF   256
#define EDGES 131072
#define NNODE (N_PED*PPATCH)   // 16384
#define SEQ   (LPATCH+PRED)    // 15
#define ROWS  (N_PED*SEQ)      // 61440

typedef __attribute__((ext_vector_type(16))) __bf16         v16bf;
typedef __attribute__((ext_vector_type(16))) unsigned short v16us;
typedef __attribute__((ext_vector_type(8)))  float          v8f;

// ---------------- helpers ----------------
__device__ __forceinline__ unsigned short f2bf(float f) {
    unsigned u = __float_as_uint(f);
    u += 0x7fffu + ((u >> 16) & 1u);        // round-to-nearest-even
    return (unsigned short)(u >> 16);
}
__device__ __forceinline__ unsigned fenc(float f) {     // order-preserving float->uint
    unsigned u = __float_as_uint(f);
    return (u & 0x80000000u) ? ~u : (u | 0x80000000u);
}
__device__ __forceinline__ float fdec(unsigned k) {
    unsigned u = (k & 0x80000000u) ? (k & 0x7fffffffu) : ~k;
    return __uint_as_float(u);
}

// =====================================================================
// WMMA GEMM:  C[M,N] = act( A[M,K] @ B[K,N] + bias )
// A fp32 row-major (converted to bf16 on load), B pre-packed bf16 fragments.
// Each wave computes a 16x128 strip = 8 C tiles: the A fragment (and its
// fp32->bf16 conversion) is loaded once per k-step and reused by 8
// v_wmma_f32_16x16x32_bf16 issues. 4 waves/block cover 64 rows.
// act: 0 = none, 1 = tanh-gelu
// =====================================================================
#define GEMM_NT 8    // column tiles per wave (128 columns)

__global__ __launch_bounds__(128)
void gemm_bf16_wmma(const float* __restrict__ A, const unsigned short* __restrict__ Bp,
                    const float* __restrict__ bias, float* __restrict__ C,
                    int M, int N, int K, int act)
{
    const int lane = threadIdx.x & 31;
    const int wave = threadIdx.x >> 5;
    const int mt = blockIdx.x * 4 + wave;          // 16-row tile index
    const int ct0 = blockIdx.y * GEMM_NT;          // first 16-col tile index
    const int nt = N >> 4;

    const int row = mt * 16 + (lane & 15);
    const int kh  = lane >> 4;                     // which K-half this lane holds
    const float* ap = A + (size_t)row * K + kh * 16;
    const unsigned short* bp = Bp + ((size_t)ct0 * 32 + lane) * 16;
    const size_t bstep = (size_t)nt * 512;         // ushorts per k-step of packed B
    const int ktn = K >> 5;

    v8f acc[GEMM_NT] = {};
    for (int kt = 0; kt < ktn; ++kt) {
        __builtin_prefetch(ap + 64, 0, 1);         // global_prefetch_b8 on A stream
        v16us au;
        #pragma unroll
        for (int e = 0; e < 8; ++e) au[e]     = f2bf(ap[e]);       // K = base..base+7
        #pragma unroll
        for (int e = 0; e < 8; ++e) au[8 + e] = f2bf(ap[8 + e]);   // K = base+8..base+15
        v16bf af = __builtin_bit_cast(v16bf, au);
        #pragma unroll
        for (int t = 0; t < GEMM_NT; ++t) {
            v16bf bf = *(const v16bf*)(bp + (size_t)t * 512);      // contiguous 32B fragment
            acc[t] = __builtin_amdgcn_wmma_f32_16x16x32_bf16(false, af, false, bf,
                                                             (short)0, acc[t], false, false);
        }
        ap += 32;
        bp += bstep;
    }

    const int gr = mt * 16 + kh * 8;
    #pragma unroll
    for (int t = 0; t < GEMM_NT; ++t) {
        const int gc = (ct0 + t) * 16 + (lane & 15);
        const float bv = bias ? bias[gc] : 0.f;
        #pragma unroll
        for (int r = 0; r < 8; ++r) {
            float v = acc[t][r] + bv;
            if (act == 1) {
                float tt = 0.7978845608028654f * (v + 0.044715f * v * v * v);
                v = 0.5f * v * (1.f + tanhf(tt));
            }
            C[(size_t)(gr + r) * N + gc] = v;
        }
    }
}

// Pack fp32 weight W[K,N] into bf16 WMMA B-fragment order:
//   frag[((kt*nt+ct)*32+lane)*16+e] = bf16( W[kt*32+(lane>>4)*16+e][ct*16+(lane&15)] )
__global__ void pack_weight_bf16(const float* __restrict__ W, unsigned short* __restrict__ out,
                                 int K, int N)
{
    int tid = blockIdx.x * blockDim.x + threadIdx.x;
    int nt = N >> 4;
    int total = (K >> 5) * nt * 32;
    if (tid >= total) return;
    int lane = tid & 31;
    int t2 = tid >> 5;
    int ct = t2 % nt, kt = t2 / nt;
    int col = ct * 16 + (lane & 15);
    int kbase = kt * 32 + ((lane >> 4) << 4);
    unsigned short* o = out + (size_t)tid * 16;
    #pragma unroll
    for (int e = 0; e < 16; ++e)
        o[e] = f2bf(W[(size_t)(kbase + e) * N + col]);
}

// =====================================================================
// Hodge collapse: enorm[e,i] = ea[e]*s_i + t_i   (K=1 Laguerre => affine in ea)
// =====================================================================
__global__ void hodge_consts_kernel(const float* Wee, const float* bee,
                                    const float* Whod, const float* bhod,
                                    const float* Wec, const float* bec, float* consts)
{
    __shared__ float u[DIM], c[DIM];
    int j = threadIdx.x;
    float us = 0.f, cs = 0.f;
    for (int k = 0; k < DIM; ++k) {
        float w = Whod[k * DIM + j];
        us += Wee[k] * w;
        cs += bee[k] * w;
    }
    u[j] = us; c[j] = cs + bhod[j];
    __syncthreads();
    if (j < 2) {
        float s = 0.f, t = 0.f;
        for (int q = 0; q < DIM; ++q) { s += u[q] * Wec[q * 2 + j]; t += c[q] * Wec[q * 2 + j]; }
        consts[j] = s; consts[2 + j] = t + bec[j];
    }
}

// ---------------- front-end features (rank-1 embeddings) ----------------
__global__ void feat_kernel(const float* __restrict__ vel, const float* __restrict__ nodes,
                            const float* Wn, const float* bn, const float* Wa, const float* ba,
                            const float* Wv, const float* bv, float* __restrict__ feats)
{
    int ntid = blockIdx.x;             // n*OBS + t
    int n = ntid / OBS, t = ntid % OBS;
    int d = threadIdx.x;
    float vx = vel[(size_t)(n * OBS + t) * 2 + 0];
    float vy = vel[(size_t)(n * OBS + t) * 2 + 1];
    int tp = (t == 0) ? 0 : t - 1;
    float px = vel[(size_t)(n * OBS + tp) * 2 + 0];
    float py = vel[(size_t)(n * OBS + tp) * 2 + 1];
    const float EPS = 1e-6f;
    float cosv = (px * vx + py * vy) /
                 ((sqrtf(px * px + py * py) + EPS) * (sqrtf(vx * vx + vy * vy) + EPS));
    cosv = fminf(1.f, fmaxf(-1.f, cosv));
    float ang = acosf(cosv);
    float out;
    if (d < FEATW)            out = nodes[n * OBS + t] * Wn[d] + bn[d];
    else if (d < 2 * FEATW) { int j = d - FEATW;     out = ang * Wa[j] + ba[j]; }
    else                    { int j = d - 2 * FEATW; out = vx * Wv[j] + vy * Wv[64 + j] + bv[j]; }
    feats[(size_t)ntid * DIM + d] = out;
}

__global__ void build_z1_kernel(const float* __restrict__ feats, float* __restrict__ z1)
{
    int idx = blockIdx.x;              // l*NNODE + nn
    int d = threadIdx.x;
    int l = idx / NNODE, nn = idx % NNODE;
    int n = nn >> 2, p = nn & 3;
    int t = l * PSTRIDE + p;
    z1[(size_t)idx * DIM + d] = feats[(size_t)(n * OBS + t) * DIM + d];
}

// ---------------- GAT / GCN graph stage ----------------
__global__ void init_node_acc_kernel(const float* bgat, float* node, float* acc)
{
    int i = blockIdx.x, d = threadIdx.x;
    node[(size_t)i * DIM + d] = bgat[d];   // segment_sum + b_gat baked into init
    acc[(size_t)i * DIM + d] = 0.f;
}
__global__ void init_ms_kernel(unsigned* mkey, float* ssum)
{
    int i = blockIdx.x * blockDim.x + threadIdx.x;
    if (i < NNODE) { mkey[i] = 0u; ssum[i] = 0.f; }
}

__global__ __launch_bounds__(256)
void gat_score_kernel(const int* __restrict__ ei, const float* __restrict__ xl,
                      const float* __restrict__ xr, const float* __restrict__ agat,
                      float* __restrict__ esc, unsigned* __restrict__ mkey)
{
    int lane = threadIdx.x & 31, wv = threadIdx.x >> 5;
    int e = blockIdx.x * 8 + wv;
    if (e >= EDGES) return;
    int src = ei[e], dst = ei[EDGES + e];
    float s = 0.f;
    #pragma unroll
    for (int q = 0; q < 4; ++q) {
        int d = lane * 4 + q;
        float x = xl[(size_t)src * DIM + d] + xr[(size_t)dst * DIM + d];
        x = x > 0.f ? x : 0.2f * x;                 // leaky_relu(0.2)
        s += x * agat[d];
    }
    for (int off = 16; off; off >>= 1) s += __shfl_xor(s, off, 32);
    if (lane == 0) { esc[e] = s; atomicMax(&mkey[dst], fenc(s)); }
}

__global__ void gat_exp_kernel(const int* __restrict__ ei, const float* __restrict__ esc,
                               const unsigned* __restrict__ mkey, float* __restrict__ exv,
                               float* __restrict__ ssum)
{
    int e = blockIdx.x * blockDim.x + threadIdx.x;
    if (e >= EDGES) return;
    int dst = ei[EDGES + e];
    float ex = expf(esc[e] - fdec(mkey[dst]));
    exv[e] = ex;
    atomicAdd(&ssum[dst], ex);
}

__global__ __launch_bounds__(256)
void gat_scatter_kernel(const int* __restrict__ ei, const float* __restrict__ exv,
                        const float* __restrict__ ssum, const float* __restrict__ xl,
                        float* __restrict__ node)
{
    int lane = threadIdx.x & 31, wv = threadIdx.x >> 5;
    int e = blockIdx.x * 8 + wv;
    if (e >= EDGES) return;
    int src = ei[e], dst = ei[EDGES + e];
    float alpha = exv[e] / (ssum[dst] + 1e-16f);
    #pragma unroll
    for (int q = 0; q < 4; ++q) {
        int d = lane * 4 + q;
        atomicAdd(&node[(size_t)dst * DIM + d], alpha * xl[(size_t)src * DIM + d]);
    }
}

__global__ __launch_bounds__(256)
void gat_mfmb_kernel(const int* __restrict__ ei, const float* __restrict__ ea,
                     const float* __restrict__ consts, const float* __restrict__ node,
                     float* __restrict__ acc)
{
    int lane = threadIdx.x & 31, wv = threadIdx.x >> 5;
    int e = blockIdx.x * 8 + wv;
    if (e >= EDGES) return;
    int src = ei[e], dst = ei[EDGES + e];
    float a = ea[e];
    unsigned ua = __float_as_uint(a);
    if ((ua & 0x7f800000u) == 0x7f800000u) a = 0.f;   // nan/inf -> 0
    float e0 = a * consts[0] + consts[2];
    float e1 = a * consts[1] + consts[3];
    #pragma unroll
    for (int q = 0; q < 4; ++q) {
        int d = lane * 4 + q;
        atomicAdd(&acc[(size_t)dst * DIM + d], e0 * node[(size_t)src * DIM + d]);
        atomicAdd(&acc[(size_t)src * DIM + d], e1 * node[(size_t)dst * DIM + d]);
    }
}

// traj[n,l,:] = mean_p( node2[n*4+p] + z1[n*4+p] )   written into x[:, l, :]
__global__ void patch_reduce_kernel(const float* __restrict__ node2, const float* __restrict__ z1l,
                                    float* __restrict__ x, int l)
{
    int n = blockIdx.x, d = threadIdx.x;
    float s = 0.f;
    #pragma unroll
    for (int p = 0; p < 4; ++p) {
        size_t idx = (size_t)(n * 4 + p) * DIM + d;
        s += node2[idx] + z1l[idx];
    }
    x[((size_t)n * SEQ + l) * DIM + d] = 0.25f * s;
}

__global__ void assemble_x_kernel(const float* __restrict__ step_ph, const float* __restrict__ Wpos,
                                  float* __restrict__ x)
{
    int ns = blockIdx.x;               // n*SEQ + s
    int n = ns / SEQ, s = ns % SEQ, d = threadIdx.x;
    size_t idx = (size_t)ns * DIM + d;
    float wp = Wpos[s * DIM + d];
    if (s < LPATCH) x[idx] += wp;
    else            x[idx] = step_ph[(size_t)n * DIM + d] + wp;
}

// ---------------- transformer pieces ----------------
__global__ __launch_bounds__(128)
void attn_kernel(const float* __restrict__ qb, const float* __restrict__ kb,
                 const float* __restrict__ vb, float* __restrict__ ob)
{
    int tid = blockIdx.x * blockDim.x + threadIdx.x;
    const int total = N_PED * NHEAD * SEQ;
    if (tid >= total) return;
    int s = tid % SEQ;
    int t2 = tid / SEQ;
    int h = t2 % NHEAD, n = t2 / NHEAD;
    const size_t base = (size_t)(n * SEQ) * DIM + h * 32;

    float q[32];
    #pragma unroll
    for (int i = 0; i < 32; ++i) q[i] = qb[base + (size_t)s * DIM + i];
    float sc[SEQ];
    float mx = -3.402823466e38f;
    #pragma unroll
    for (int t = 0; t < SEQ; ++t) {
        float d = 0.f;
        #pragma unroll
        for (int i = 0; i < 32; ++i) d += q[i] * kb[base + (size_t)t * DIM + i];
        d *= 0.17677669529663687f;      // 1/sqrt(32)
        sc[t] = d; mx = fmaxf(mx, d);
    }
    float den = 0.f;
    #pragma unroll
    for (int t = 0; t < SEQ; ++t) { sc[t] = expf(sc[t] - mx); den += sc[t]; }
    float inv = 1.f / den;
    float o[32];
    #pragma unroll
    for (int i = 0; i < 32; ++i) o[i] = 0.f;
    #pragma unroll
    for (int t = 0; t < SEQ; ++t) {
        float w = sc[t] * inv;
        #pragma unroll
        for (int i = 0; i < 32; ++i) o[i] += w * vb[base + (size_t)t * DIM + i];
    }
    #pragma unroll
    for (int i = 0; i < 32; ++i) ob[base + (size_t)s * DIM + i] = o[i];
}

// out = LayerNorm(a + b) ; b may be null
__global__ __launch_bounds__(256)
void add_ln_kernel(const float* __restrict__ a, const float* __restrict__ b,
                   const float* __restrict__ g, const float* __restrict__ bb,
                   float* __restrict__ out)
{
    int lane = threadIdx.x & 31, wv = threadIdx.x >> 5;
    int row = blockIdx.x * 8 + wv;
    if (row >= ROWS) return;
    size_t base = (size_t)row * DIM;
    float v[4], s = 0.f;
    #pragma unroll
    for (int q = 0; q < 4; ++q) {
        int d = lane * 4 + q;
        float x = a[base + d];
        if (b) x += b[base + d];
        v[q] = x; s += x;
    }
    for (int off = 16; off; off >>= 1) s += __shfl_xor(s, off, 32);
    float m = s * (1.f / DIM);
    float var = 0.f;
    #pragma unroll
    for (int q = 0; q < 4; ++q) { float dd = v[q] - m; var += dd * dd; }
    for (int off = 16; off; off >>= 1) var += __shfl_xor(var, off, 32);
    float rs = rsqrtf(var * (1.f / DIM) + 1e-5f);
    #pragma unroll
    for (int q = 0; q < 4; ++q) {
        int d = lane * 4 + q;
        out[base + d] = (v[q] - m) * rs * g[d] + bb[d];
    }
}

__global__ __launch_bounds__(256)
void out_proj_kernel(const float* __restrict__ xf, const float* __restrict__ Wo,
                     const float* __restrict__ bo, float* __restrict__ out)
{
    int lane = threadIdx.x & 31, wv = threadIdx.x >> 5;
    int row = blockIdx.x * 8 + wv;
    if (row >= ROWS) return;
    float p[5] = {0.f, 0.f, 0.f, 0.f, 0.f};
    #pragma unroll
    for (int q = 0; q < 4; ++q) {
        int d = lane + 32 * q;
        float xv = xf[(size_t)row * DIM + d];
        #pragma unroll
        for (int j = 0; j < 5; ++j) p[j] += xv * Wo[d * 5 + j];
    }
    #pragma unroll
    for (int j = 0; j < 5; ++j)
        for (int off = 16; off; off >>= 1) p[j] += __shfl_xor(p[j], off, 32);
    if (lane == 0) {
        #pragma unroll
        for (int j = 0; j < 5; ++j) out[(size_t)row * 5 + j] = p[j] + bo[j];
    }
}

// =====================================================================
// host-side orchestration
// =====================================================================
static inline void launch_gemm(const float* A, const unsigned short* Bp, const float* bias,
                               float* C, int M, int N, int K, int act, hipStream_t s)
{
    dim3 g(M / 64, N / 128);           // 4 row-tiles per block, 8 col-tiles per wave
    gemm_bf16_wmma<<<g, 128, 0, s>>>(A, Bp, bias, C, M, N, K, act);
}
static inline void launch_pack(const float* W, unsigned short* out, int K, int N, hipStream_t s)
{
    int total = (K / 32) * (N / 16) * 32;
    pack_weight_bf16<<<(total + 255) / 256, 256, 0, s>>>(W, out, K, N);
}

// workspace layout (bytes)
#define OFF_CONST  ((size_t)0)
#define OFF_PACK   ((size_t)256)
#define PK_GATL    ((size_t)0)
#define PK_GATR    ((size_t)16384)
#define PK_GCN     ((size_t)32768)
#define PK_LAYER(l) ((size_t)49152 + (size_t)(l) * 131072)   // Wq,Wk,Wv,Wo(16384 each),W1(32768),W2(32768)
#define PACK_BYTES ((size_t)(49152 + 2 * 131072) * 2)
#define OFF_FEATS  (OFF_PACK + PACK_BYTES)
#define OFF_Z1     (OFF_FEATS + (size_t)N_PED * OBS * DIM * 4)
#define OFF_XL     (OFF_Z1 + (size_t)LPATCH * NNODE * DIM * 4)
#define OFF_XR     (OFF_XL + (size_t)NNODE * DIM * 4)
#define OFF_NODE   (OFF_XR + (size_t)NNODE * DIM * 4)
#define OFF_ACC    (OFF_NODE + (size_t)NNODE * DIM * 4)
#define OFF_NODE2  (OFF_ACC + (size_t)NNODE * DIM * 4)
#define OFF_ESC    (OFF_NODE2 + (size_t)NNODE * DIM * 4)
#define OFF_EX     (OFF_ESC + (size_t)EDGES * 4)
#define OFF_MKEY   (OFF_EX + (size_t)EDGES * 4)
#define OFF_SSUM   (OFF_MKEY + (size_t)NNODE * 4)
#define OFF_X      (OFF_SSUM + (size_t)NNODE * 4)
#define OFF_QB     (OFF_X + (size_t)ROWS * DIM * 4)
#define OFF_KB     (OFF_QB + (size_t)ROWS * DIM * 4)
#define OFF_VB     (OFF_KB + (size_t)ROWS * DIM * 4)
#define OFF_OB     (OFF_VB + (size_t)ROWS * DIM * 4)
#define WS_NEEDED  (OFF_OB + (size_t)ROWS * DIM * 4)          // ~243 MB

extern "C" void kernel_launch(void* const* d_in, const int* in_sizes, int n_in,
                              void* d_out, int out_size, void* d_ws, size_t ws_size,
                              hipStream_t stream)
{
    if (ws_size < WS_NEEDED) return;   // workspace too small: safe no-op

    const float* vel        = (const float*)d_in[1];
    const float* nodes      = (const float*)d_in[2];
    const int*   edge_index = (const int*)d_in[3];
    const float* edge_attr  = (const float*)d_in[4];
    // params (d_in[6..61]) in make_params insertion order
#define PAR(i) ((const float*)d_in[6 + (i)])
    const float *W_norm = PAR(0), *b_norm = PAR(1), *W_angle = PAR(2), *b_angle = PAR(3);
    const float *W_vel = PAR(4), *b_vel = PAR(5), *W_ee = PAR(6), *b_ee = PAR(7);
    const float *W_hodge = PAR(8), *b_hodge = PAR(9), *W_ec = PAR(10), *b_ec = PAR(11);
    const float *W_gcn = PAR(12), *b_gcn = PAR(13), *W_out = PAR(14), *b_out = PAR(15);
    const float *W_gat_l = PAR(16), *W_gat_r = PAR(17), *a_gat = PAR(18), *b_gat = PAR(19);
    const float *step_ph = PAR(20), *W_pos = PAR(21), *ln_f_g = PAR(22), *ln_f_b = PAR(23);
    // layer i base: 24 + 16*i : Wq,bq,Wk,bk,Wv,bv,Wo,bo,W1,b1,W2,b2,ln1_g,ln1_b,ln2_g,ln2_b

    char* ws = (char*)d_ws;
    float* consts = (float*)(ws + OFF_CONST);
    unsigned short* pk = (unsigned short*)(ws + OFF_PACK);
    float* feats = (float*)(ws + OFF_FEATS);
    float* z1    = (float*)(ws + OFF_Z1);
    float* xl    = (float*)(ws + OFF_XL);
    float* xr    = (float*)(ws + OFF_XR);
    float* node  = (float*)(ws + OFF_NODE);
    float* acc   = (float*)(ws + OFF_ACC);
    float* node2 = (float*)(ws + OFF_NODE2);
    float* esc   = (float*)(ws + OFF_ESC);
    float* exv   = (float*)(ws + OFF_EX);
    unsigned* mkey = (unsigned*)(ws + OFF_MKEY);
    float* ssum  = (float*)(ws + OFF_SSUM);
    float* x     = (float*)(ws + OFF_X);
    float* qb    = (float*)(ws + OFF_QB);
    float* kb    = (float*)(ws + OFF_KB);
    float* vb    = (float*)(ws + OFF_VB);
    float* ob    = (float*)(ws + OFF_OB);

    // ---- constants + weight packing ----
    hodge_consts_kernel<<<1, DIM, 0, stream>>>(W_ee, b_ee, W_hodge, b_hodge, W_ec, b_ec, consts);
    launch_pack(W_gat_l, pk + PK_GATL, DIM, DIM, stream);
    launch_pack(W_gat_r, pk + PK_GATR, DIM, DIM, stream);
    launch_pack(W_gcn,   pk + PK_GCN,  DIM, DIM, stream);
    for (int l = 0; l < 2; ++l) {
        int pb = 24 + 16 * l;
        unsigned short* base = pk + PK_LAYER(l);
        launch_pack(PAR(pb + 0),  base + 0,      DIM, DIM, stream);  // Wq
        launch_pack(PAR(pb + 2),  base + 16384,  DIM, DIM, stream);  // Wk
        launch_pack(PAR(pb + 4),  base + 32768,  DIM, DIM, stream);  // Wv
        launch_pack(PAR(pb + 6),  base + 49152,  DIM, DIM, stream);  // Wo
        launch_pack(PAR(pb + 8),  base + 65536,  DIM, DFF, stream);  // W1
        launch_pack(PAR(pb + 10), base + 98304,  DFF, DIM, stream);  // W2
    }

    // ---- front end ----
    feat_kernel<<<N_PED * OBS, DIM, 0, stream>>>(vel, nodes, W_norm, b_norm, W_angle, b_angle,
                                                 W_vel, b_vel, feats);
    build_z1_kernel<<<LPATCH * NNODE, DIM, 0, stream>>>(feats, z1);

    // ---- per-patch graph stage ----
    for (int l = 0; l < LPATCH; ++l) {
        const int* ei = edge_index + (size_t)l * 2 * EDGES;
        const float* ea = edge_attr + (size_t)l * EDGES;
        const float* z1l = z1 + (size_t)l * NNODE * DIM;
        launch_gemm(z1l, pk + PK_GATL, nullptr, xl, NNODE, DIM, DIM, 0, stream);
        launch_gemm(z1l, pk + PK_GATR, nullptr, xr, NNODE, DIM, DIM, 0, stream);
        init_node_acc_kernel<<<NNODE, DIM, 0, stream>>>(b_gat, node, acc);
        init_ms_kernel<<<NNODE / 256, 256, 0, stream>>>(mkey, ssum);
        gat_score_kernel<<<EDGES / 8, 256, 0, stream>>>(ei, xl, xr, a_gat, esc, mkey);
        gat_exp_kernel<<<EDGES / 256, 256, 0, stream>>>(ei, esc, mkey, exv, ssum);
        gat_scatter_kernel<<<EDGES / 8, 256, 0, stream>>>(ei, exv, ssum, xl, node);
        gat_mfmb_kernel<<<EDGES / 8, 256, 0, stream>>>(ei, ea, consts, node, acc);
        launch_gemm(acc, pk + PK_GCN, b_gcn, node2, NNODE, DIM, DIM, 0, stream);
        patch_reduce_kernel<<<N_PED, DIM, 0, stream>>>(node2, z1l, x, l);
    }
    assemble_x_kernel<<<N_PED * SEQ, DIM, 0, stream>>>(step_ph, W_pos, x);

    // ---- transformer (2 layers) ----
    for (int l = 0; l < 2; ++l) {
        int pb = 24 + 16 * l;
        unsigned short* base = pk + PK_LAYER(l);
        launch_gemm(x, base + 0,     PAR(pb + 1), qb, ROWS, DIM, DIM, 0, stream);   // q
        launch_gemm(x, base + 16384, PAR(pb + 3), kb, ROWS, DIM, DIM, 0, stream);   // k
        launch_gemm(x, base + 32768, PAR(pb + 5), vb, ROWS, DIM, DIM, 0, stream);   // v
        attn_kernel<<<(N_PED * NHEAD * SEQ) / 128, 128, 0, stream>>>(qb, kb, vb, ob);
        launch_gemm(ob, base + 49152, PAR(pb + 7), qb /*o2*/, ROWS, DIM, DIM, 0, stream);
        add_ln_kernel<<<ROWS / 8, 256, 0, stream>>>(x, qb, PAR(pb + 12), PAR(pb + 13), kb /*x2*/);
        launch_gemm(kb, base + 65536, PAR(pb + 9), vb /*h1 spans vb+ob*/, ROWS, DFF, DIM, 1, stream);
        launch_gemm(vb, base + 98304, PAR(pb + 11), qb /*y*/, ROWS, DIM, DFF, 0, stream);
        add_ln_kernel<<<ROWS / 8, 256, 0, stream>>>(kb, qb, PAR(pb + 14), PAR(pb + 15), x);
    }

    // ---- final LN + output projection ----
    add_ln_kernel<<<ROWS / 8, 256, 0, stream>>>(x, nullptr, ln_f_g, ln_f_b, qb);
    out_proj_kernel<<<ROWS / 8, 256, 0, stream>>>(qb, W_out, b_out, (float*)d_out);
#undef PAR
}
